// MultiHeadAttentionQuantum_65481071401117
// MI455X (gfx1250) — matrix operand order repaired and checked by reading, not compiled, not run on hardware
//
#include <hip/hip_runtime.h>
#include <cstdint>

typedef __attribute__((ext_vector_type(16))) _Float16 v16h;
typedef __attribute__((ext_vector_type(8)))  float    v8f;

#define BB     2
#define SS     2048
#define EMB    1024
#define NH     16
#define DKK    64

#if __has_builtin(__builtin_amdgcn_exp2f)
#define EXP2F(x) __builtin_amdgcn_exp2f(x)
#else
#define EXP2F(x) exp2f(x)
#endif

__device__ __forceinline__ v8f wmma_f16(v16h a, v16h b, v8f c) {
  // D = A(16x32 f16) x B(32x16 f16) + C(16x16 f32)
  return __builtin_amdgcn_wmma_f32_16x16x32_f16(false, a, false, b, (short)0, c, false, false);
}

// ---- CDNA5 async global->LDS copy (tracked by ASYNCcnt) --------------------
__device__ __forceinline__ void async_b128(uint32_t lds_addr, const void* gptr) {
  uint64_t ga = (uint64_t)gptr;
  asm volatile("global_load_async_to_lds_b128 %0, %1, off"
               :: "v"(lds_addr), "v"(ga) : "memory");
}
__device__ __forceinline__ void wait_async0() {
  asm volatile("s_wait_asynccnt 0x0" ::: "memory");
}
// issue one 4KB chunk (256 x b128) with 128 threads -> 2 per thread
__device__ __forceinline__ void stage4k(const void* src, void* dstLds, int tid) {
  uint32_t l0 = (uint32_t)(uintptr_t)dstLds + (uint32_t)tid * 16u;
  const char* g = (const char*)src + tid * 16;
  async_b128(l0, g);
  async_b128(l0 + 2048u, g + 2048);
}

// ---------------------------------------------------------------------------
// Kernel 1: quantum heads. c = cos(x + theta); out0 = prod_{i>=1} c_i,
// out_k = cumprod_k for k>=1. Write f16 q in [b, h, s, d] layout.
// ---------------------------------------------------------------------------
__global__ __launch_bounds__(256)
void qgen_kernel(const float* __restrict__ x, const float* __restrict__ theta,
                 _Float16* __restrict__ q16) {
  int g = blockIdx.x * blockDim.x + threadIdx.x;      // (b, s, h) flattened
  if (g >= BB * SS * NH) return;
  int h = g % NH;
  int s = (g / NH) % SS;
  int b = g / (NH * SS);
  const float* xp = x + ((size_t)(b * SS + s)) * EMB + h * DKK;
  _Float16* qp = q16 + (((size_t)(b * NH + h)) * SS + s) * DKK;

  float cp = __cosf(xp[0] + theta[0]);
  float p1 = 1.0f;
  #pragma unroll 8
  for (int i = 1; i < DKK; ++i) {
    float ci = __cosf(xp[i] + theta[i]);
    cp *= ci;
    p1 *= ci;
    qp[i] = (_Float16)cp;
  }
  qp[0] = (_Float16)p1;
}

// ---------------------------------------------------------------------------
// Kernel 2: w_out f32 -> f16
// ---------------------------------------------------------------------------
__global__ __launch_bounds__(256)
void wconv_kernel(const float* __restrict__ w, _Float16* __restrict__ w16, int n) {
  int i = blockIdx.x * blockDim.x + threadIdx.x;
  if (i < n) w16[i] = (_Float16)w[i];
}

// ---------------------------------------------------------------------------
// Kernel 3: attention, q == k == v. Unnormalized exp accumulation (scores are
// bounded by |q.k|/8 <= 8, so no max-stabilization needed in f32).
// Block = (b, h, 128-query block); 4 waves, each owns TWO 16-query tiles so
// every B fragment feeds two WMMAs. Double-buffered async K/V staging.
// ---------------------------------------------------------------------------
__global__ __launch_bounds__(128)
void attn_kernel(const _Float16* __restrict__ q16, _Float16* __restrict__ o16) {
  __shared__ __align__(16) _Float16 kv[2][32 * DKK];  // 2 x (32 keys x 64 dims)
  __shared__ _Float16 plds[4][32 * 32];               // per-wave P tile (32q x 32k)

  const int tid  = threadIdx.x;
  const int lane = tid & 31;
  const int wave = tid >> 5;
  const int hv   = (lane >= 16) ? 1 : 0;              // lane half
  const int ln   = lane & 15;

  const int QB   = SS / 128;
  const int qblk = blockIdx.x % QB;
  const int bh   = blockIdx.x / QB;                   // b*NH + h
  const _Float16* qhead = q16 + (size_t)bh * SS * DKK;
  const int qbase = qblk * 128 + wave * 32;           // 32 queries per wave

  // Q A-fragments: 2 tiles x (16 rows x 64 dims -> two 16x32 f16 fragments)
  v16h aQ0[2], aQ1[2];
  #pragma unroll
  for (int t = 0; t < 2; ++t) {
    const _Float16* qrow = qhead + (size_t)(qbase + t * 16 + ln) * DKK;
    #pragma unroll
    for (int j = 0; j < 16; ++j) {
      int k = j + (hv ? 16 : 0);
      aQ0[t][j] = qrow[k];
      aQ1[t][j] = qrow[32 + k];
    }
  }

  float lacc[2][8];
  v8f O[2][4];
  #pragma unroll
  for (int t = 0; t < 2; ++t) {
    #pragma unroll
    for (int r = 0; r < 8; ++r) lacc[t][r] = 0.0f;
    #pragma unroll
    for (int dt = 0; dt < 4; ++dt) O[t][dt] = (v8f){0, 0, 0, 0, 0, 0, 0, 0};
  }

  const float SCL2 = 0.18033688011112042f;            // (1/8) * log2(e)

  stage4k(qhead, kv[0], tid);                         // prefetch chunk 0

  for (int kc = 0; kc < SS; kc += 32) {
    const int cur = (kc >> 5) & 1;
    wait_async0();                                    // my async loads landed
    __syncthreads();                                  // buffer published, prev reads done
    if (kc + 32 < SS)
      stage4k(qhead + (size_t)(kc + 32) * DKK, kv[cur ^ 1], tid);

    const _Float16* kvc = kv[cur];

    // Scores: c[t][u] = Qtile_t x Ktile_u, each B fragment used twice
    v8f c[2][2];
    {
      v16h b00, b01, b10, b11;
      #pragma unroll
      for (int j = 0; j < 16; ++j) {
        int d = j + (hv ? 16 : 0);                    // B frag K index = feature dim
        b00[j] = kvc[ln * DKK + d];
        b01[j] = kvc[ln * DKK + 32 + d];
        b10[j] = kvc[(16 + ln) * DKK + d];
        b11[j] = kvc[(16 + ln) * DKK + 32 + d];
      }
      #pragma unroll
      for (int t = 0; t < 2; ++t) {
        c[t][0] = (v8f){0, 0, 0, 0, 0, 0, 0, 0};
        c[t][1] = (v8f){0, 0, 0, 0, 0, 0, 0, 0};
        c[t][0] = wmma_f16(aQ0[t], b00, c[t][0]);
        c[t][0] = wmma_f16(aQ1[t], b01, c[t][0]);
        c[t][1] = wmma_f16(aQ0[t], b10, c[t][1]);
        c[t][1] = wmma_f16(aQ1[t], b11, c[t][1]);
      }
    }

    // Unnormalized softmax weights; accumulate per-lane row-sum partials
    #pragma unroll
    for (int t = 0; t < 2; ++t) {
      #pragma unroll
      for (int r = 0; r < 8; ++r) {
        float p0 = EXP2F(c[t][0][r] * SCL2);
        float p1 = EXP2F(c[t][1][r] * SCL2);
        lacc[t][r] += p0 + p1;
        int row = t * 16 + r + hv * 8;
        plds[wave][row * 32 + ln]      = (_Float16)p0;
        plds[wave][row * 32 + 16 + ln] = (_Float16)p1;
      }
    }

    __builtin_amdgcn_wave_barrier();   // order per-wave LDS write->read

    // Re-shape P into A-fragments and accumulate P.V (V frags shared by tiles)
    v16h aP[2];
    #pragma unroll
    for (int t = 0; t < 2; ++t)
      #pragma unroll
      for (int j = 0; j < 16; ++j) {
        int k = j + (hv ? 16 : 0);
        aP[t][j] = plds[wave][(t * 16 + ln) * 32 + k];
      }
    #pragma unroll
    for (int dt = 0; dt < 4; ++dt) {
      v16h bV;
      #pragma unroll
      for (int j = 0; j < 16; ++j) {
        int kk = j + (hv ? 16 : 0);                   // key index within chunk
        bV[j] = kvc[kk * DKK + dt * 16 + ln];
      }
      O[0][dt] = wmma_f16(aP[0], bV, O[0][dt]);
      O[1][dt] = wmma_f16(aP[1], bV, O[1][dt]);
    }
  }

  // One final row-sum reduction across the 16 lanes of each half
  const int b = bh / NH, h = bh % NH;
  #pragma unroll
  for (int t = 0; t < 2; ++t) {
    #pragma unroll
    for (int r = 0; r < 8; ++r) {
      float s = lacc[t][r];
      s += __shfl_xor(s, 1, 32);
      s += __shfl_xor(s, 2, 32);
      s += __shfl_xor(s, 4, 32);
      s += __shfl_xor(s, 8, 32);
      float linv = 1.0f / s;
      int qi = qbase + t * 16 + r + hv * 8;
      #pragma unroll
      for (int dt = 0; dt < 4; ++dt) {
        float v = O[t][dt][r] * linv;
        o16[((size_t)(b * SS + qi)) * EMB + h * DKK + dt * 16 + ln] = (_Float16)v;
      }
    }
  }
}

// ---------------------------------------------------------------------------
// Kernel 4: out[m, n] = sum_k a16[m, k] * w16[n, k]   (out = attn @ w_out^T)
// Block computes a 128x64 output tile (wave = 32 rows x 64 cols, so each W
// fragment feeds two WMMAs). Double-buffered async staging.
// ---------------------------------------------------------------------------
__global__ __launch_bounds__(128)
void proj_kernel(const _Float16* __restrict__ a16, const _Float16* __restrict__ w16,
                 float* __restrict__ out) {
  __shared__ __align__(16) _Float16 aT[2][128 * 32];  // 2 x 8KB
  __shared__ __align__(16) _Float16 wT[2][64 * 32];   // 2 x 4KB

  const int tid  = threadIdx.x;
  const int lane = tid & 31;
  const int wave = tid >> 5;
  const int hv   = (lane >= 16) ? 1 : 0;
  const int ln   = lane & 15;

  const int mtile = blockIdx.x >> 4;                  // 32 row tiles of 128
  const int ntile = blockIdx.x & 15;                  // 16 col tiles of 64
  const int mbase = mtile * 128;
  const int nbase = ntile * 64;

  v8f C[2][4];
  #pragma unroll
  for (int mt = 0; mt < 2; ++mt)
    #pragma unroll
    for (int dt = 0; dt < 4; ++dt) C[mt][dt] = (v8f){0, 0, 0, 0, 0, 0, 0, 0};

  // stage one k-chunk: A rows mbase..+127, W rows nbase..+63, cols [kc,kc+32)
  auto stage = [&](int kc, int buf) {
    #pragma unroll
    for (int i = 0; i < 4; ++i) {                     // 512 b128 for A
      int t   = tid * 4 + i;                          // 0..511
      int row = t >> 2;
      int off = (t & 3) * 16;
      async_b128((uint32_t)(uintptr_t)aT[buf] + row * 64 + off,
                 (const char*)(a16 + (size_t)(mbase + row) * EMB + kc) + off);
    }
    #pragma unroll
    for (int i = 0; i < 2; ++i) {                     // 256 b128 for W
      int t   = tid * 2 + i;                          // 0..255
      int row = t >> 2;
      int off = (t & 3) * 16;
      async_b128((uint32_t)(uintptr_t)wT[buf] + row * 64 + off,
                 (const char*)(w16 + (size_t)(nbase + row) * EMB + kc) + off);
    }
  };

  stage(0, 0);
  for (int kc = 0; kc < EMB; kc += 32) {
    const int cur = (kc >> 5) & 1;
    wait_async0();
    __syncthreads();
    if (kc + 32 < EMB) stage(kc + 32, cur ^ 1);

    v16h aA[2];
    #pragma unroll
    for (int mt = 0; mt < 2; ++mt)
      #pragma unroll
      for (int j = 0; j < 16; ++j) {
        int k = j + (hv ? 16 : 0);
        aA[mt][j] = aT[cur][(wave * 32 + mt * 16 + ln) * 32 + k];
      }
    #pragma unroll
    for (int dt = 0; dt < 4; ++dt) {
      v16h bW;
      #pragma unroll
      for (int j = 0; j < 16; ++j)
        bW[j] = wT[cur][(dt * 16 + ln) * 32 + j + (hv ? 16 : 0)];
      C[0][dt] = wmma_f16(aA[0], bW, C[0][dt]);
      C[1][dt] = wmma_f16(aA[1], bW, C[1][dt]);
    }
  }

  #pragma unroll
  for (int mt = 0; mt < 2; ++mt)
    #pragma unroll
    for (int dt = 0; dt < 4; ++dt)
      #pragma unroll
      for (int r = 0; r < 8; ++r)
        out[(size_t)(mbase + wave * 32 + mt * 16 + r + hv * 8) * EMB
            + nbase + dt * 16 + ln] = C[mt][dt][r];
}

// ---------------------------------------------------------------------------
extern "C" void kernel_launch(void* const* d_in, const int* in_sizes, int n_in,
                              void* d_out, int out_size, void* d_ws, size_t ws_size,
                              hipStream_t stream) {
  const float* x     = (const float*)d_in[0];
  const float* theta = (const float*)d_in[1];
  const float* w_out = (const float*)d_in[2];
  float* out = (float*)d_out;

  // Workspace layout (f16): q [B,H,S,DK] (8MB) | attn [B,S,E] (8MB) | w16 (2MB)
  _Float16* q16 = (_Float16*)d_ws;
  _Float16* a16 = q16 + (size_t)BB * NH * SS * DKK;
  _Float16* w16 = a16 + (size_t)BB * SS * EMB;

  qgen_kernel<<<(BB * SS * NH) / 256, 256, 0, stream>>>(x, theta, q16);
  wconv_kernel<<<(EMB * EMB) / 256, 256, 0, stream>>>(w_out, w16, EMB * EMB);
  attn_kernel<<<BB * NH * (SS / 128), 128, 0, stream>>>(q16, a16);
  proj_kernel<<<(BB * SS / 128) * (EMB / 64), 128, 0, stream>>>(a16, w16, out);
}
